// DiagonalSSM_62302795596586
// MI455X (gfx1250) — compile-verified, matrix operand order checked by reading, not run on hardware
//
#include <hip/hip_runtime.h>
#include <stdint.h>

// ---------------------------------------------------------------------------
// DiagonalSSM for MI455X (gfx1250, wave32, WMMA + async-LDS copies).
//   lam = sigmoid(u @ Wl^T + bl); Bu = u @ Wb^T  -> bf16 WMMA GEMM (fused 512 col)
//   h_t = lam_t * h_{t-1} + Bu_t                 -> chunked parallel scan
//   y   = H @ Wc^T + u * Dp                      -> bf16 WMMA GEMM
// GEMM operands staged in LDS via GLOBAL_LOAD_ASYNC_TO_LDS_B128, triple
// buffered (prefetch distance 2) with in-order ASYNCcnt waits + barriers.
// ---------------------------------------------------------------------------

typedef __attribute__((ext_vector_type(16))) __bf16 v16bf;
typedef __attribute__((ext_vector_type(8)))  float  v8f;

#define B_    4
#define T_    4096
#define D_    1024
#define N_    256
#define M_    (B_ * T_)       // 16384 rows
#define JG_   (2 * N_)        // 512 fused gate columns
#define CHUNK 128
#define NCH   (T_ / CHUNK)    // 32 chunks

// LDS tile: A = 32 rows x 32 k, B = 256 cols x 32 k, bf16 (elements)
#define A_ELE (32 * 32)            // 1024
#define B_ELE (256 * 32)           // 8192
#define BUF_E (A_ELE + B_ELE)      // 9216 elements = 18432 B per buffer
#define NBUF  3                    // pipeline depth (prefetch distance 2)

static __device__ __forceinline__ uint16_t f2bf(float f) {
  uint32_t x = __float_as_uint(f);
  uint32_t r = x + 0x7FFFu + ((x >> 16) & 1u);
  return (uint16_t)(r >> 16);
}

struct Frag32B { uint4 lo, hi; };

// 16x32 bf16 fragment for v_wmma_f32_16x16x32_bf16: per lane, two contiguous
// 16B groups of 8 K-elements (lanes 0-15: K[0..7],[16..23]; lanes 16-31:
// K[8..15],[24..31]).
static __device__ __forceinline__ v16bf load_frag(const uint16_t* p0,
                                                  const uint16_t* p1) {
  Frag32B r;
  r.lo = *reinterpret_cast<const uint4*>(p0);
  r.hi = *reinterpret_cast<const uint4*>(p1);
  return __builtin_bit_cast(v16bf, r);
}

// Issue one 16B async global->LDS transfer (per-lane addresses).
// vdst = 32-bit LDS byte address, vaddr = 64-bit global address.
static __device__ __forceinline__ void async_b128(uint32_t lds_addr,
                                                  const void* gp) {
  asm volatile("global_load_async_to_lds_b128 %0, %1, off"
               :: "v"(lds_addr), "v"((unsigned long long)(uintptr_t)gp)
               : "memory");
}

static __device__ __forceinline__ uint32_t lds_addr_of(const void* p) {
  // Generic (flat) LDS address: offset lives in the low 32 bits.
  return (uint32_t)(uintptr_t)p;
}

// -------------------------------- prep -------------------------------------

__global__ void __launch_bounds__(256)
cvt_f32_bf16(const float* __restrict__ src, uint16_t* __restrict__ dst, int n) {
  int i = blockIdx.x * blockDim.x + threadIdx.x;
  if (i < n) dst[i] = f2bf(src[i]);
}

__global__ void __launch_bounds__(256)
prep_wg(const float* __restrict__ Wl, const float* __restrict__ Wb,
        uint16_t* __restrict__ Wg) {
  int i = blockIdx.x * blockDim.x + threadIdx.x;   // over 512*1024
  if (i >= JG_ * D_) return;
  int j = i / D_;
  int k = i - j * D_;
  float v = (j < N_) ? Wl[j * D_ + k] : Wb[(j - N_) * D_ + k];
  Wg[i] = f2bf(v);
}

// ---------------------- shared GEMM mainloop helpers -----------------------
// Block = 256 threads (8 waves), tile M=32 x J=256, K stepped by 32.
// Wave (wm,wn): wm=wave>>2 picks 16 rows, wn=wave&3 picks 64 cols (4 wmma).

// Issue the 5 async copies (per wave) for one k-step into LDS buffer `buf`.
static __device__ __forceinline__ void issue_tile(
    uint16_t* smem, int buf, const uint16_t* __restrict__ A, int m0b,
    const uint16_t* __restrict__ W, int j0b, int K, int k, int tid) {
  uint16_t* base = smem + buf * BUF_E;
  // A tile: 32 rows x 64B; 128 16B segments (threads 128-255 mirror 0-127 so
  // every wave contributes the same ASYNCcnt=5 per step).
  {
    int ii  = tid & 127;
    int r   = ii >> 2;
    int seg = ii & 3;
    async_b128(lds_addr_of(base + r * 32 + seg * 8),
               A + (size_t)(m0b + r) * K + k + seg * 8);
  }
  // B tile: 256 rows x 64B; 1024 16B segments, 4 per thread.
#pragma unroll
  for (int q = 0; q < 4; ++q) {
    int s   = q * 256 + tid;
    int row = s >> 2;
    int sub = s & 3;
    async_b128(lds_addr_of(base + A_ELE + row * 32 + sub * 8),
               W + (size_t)(j0b + row) * K + k + sub * 8);
  }
}

// One k-step of WMMA math from LDS buffer `buf`.
// All 5 fragments are preloaded first so the 4 WMMAs issue back-to-back
// behind a single DScnt wait instead of serializing on per-pair LDS latency.
static __device__ __forceinline__ void compute_tile(
    const uint16_t* smem, int buf, int wm, int wn, int half, int mr,
    v8f acc[4]) {
  const uint16_t* As = smem + buf * BUF_E;
  const uint16_t* Bs = As + A_ELE;
  int ra = wm * 16 + mr;
  v16bf af = load_frag(As + ra * 32 + half * 8, As + ra * 32 + 16 + half * 8);
  v16bf bf[4];
#pragma unroll
  for (int t = 0; t < 4; ++t) {
    int jr = wn * 64 + t * 16 + mr;
    bf[t] = load_frag(Bs + jr * 32 + half * 8, Bs + jr * 32 + 16 + half * 8);
  }
#pragma unroll
  for (int t = 0; t < 4; ++t) {
    acc[t] = __builtin_amdgcn_wmma_f32_16x16x32_bf16(
        false, af, false, bf[t], (short)0, acc[t], false, false);
  }
}

// Triple-buffered mainloop, tail peeled so the steady-state body is
// branch-free: barrier / 5x async issue / s_wait_asynccnt 10 / barrier /
// 10x ds_load_b128 / 4x v_wmma. Requires KDIM >= 64 (both GEMMs qualify).
#define GEMM_MAINLOOP(Abase, Wbase, KDIM)                                     \
  __shared__ uint16_t smem[NBUF * BUF_E];                                     \
  const int tid  = threadIdx.x;                                               \
  const int lane = tid & 31;                                                  \
  const int wave = tid >> 5;                                                  \
  const int wm   = wave >> 2;                                                 \
  const int wn   = wave & 3;                                                  \
  const int m0b  = blockIdx.x * 32;                                           \
  const int j0b  = blockIdx.y * 256;                                          \
  const int m0   = m0b + wm * 16;                                             \
  const int j0   = j0b + wn * 64;                                             \
  const int half = lane >> 4;                                                 \
  const int mr   = lane & 15;                                                 \
  v8f acc[4] = {};                                                            \
  issue_tile(smem, 0, Abase, m0b, Wbase, j0b, KDIM, 0, tid);                  \
  issue_tile(smem, 1, Abase, m0b, Wbase, j0b, KDIM, 32, tid);                 \
  int cur = 0;                                                                \
  for (int k = 0; k + 64 < KDIM; k += 32) {                                   \
    __syncthreads(); /* buffer cur+2 safe to overwrite */                     \
    issue_tile(smem, (cur + 2) % NBUF, Abase, m0b, Wbase, j0b, KDIM,          \
               k + 64, tid);                                                  \
    asm volatile("s_wait_asynccnt 0xA" ::: "memory"); /* step k complete */   \
    __syncthreads(); /* all waves' transfers for `cur` visible */             \
    compute_tile(smem, cur, wm, wn, half, mr, acc);                           \
    cur = (cur + 1) % NBUF;                                                   \
  }                                                                           \
  /* tail: two buffers still in flight */                                     \
  asm volatile("s_wait_asynccnt 0x5" ::: "memory");                           \
  __syncthreads();                                                            \
  compute_tile(smem, cur, wm, wn, half, mr, acc);                             \
  cur = (cur + 1) % NBUF;                                                     \
  asm volatile("s_wait_asynccnt 0x0" ::: "memory");                           \
  __syncthreads();                                                            \
  compute_tile(smem, cur, wm, wn, half, mr, acc);

// ------------------------- GEMM 1: gates (fused) ---------------------------
// C[m, j] = sum_k u_bf[m,k] * Wg[j,k], M=16384, J=512, K=1024.
__global__ void __launch_bounds__(256)
gemm_gates(const uint16_t* __restrict__ Abf,   // (M_, D_)  bf16 row-major
           const uint16_t* __restrict__ Wg,    // (512, D_) bf16 row-major
           const float*    __restrict__ bl,    // (N_)
           float* __restrict__ lamBuf,         // (M_, N_)
           float* __restrict__ buBuf)          // (M_, N_)
{
  GEMM_MAINLOOP(Abf, Wg, D_)

#pragma unroll
  for (int t = 0; t < 4; ++t) {
    int j = j0 + t * 16 + mr;
#pragma unroll
    for (int r = 0; r < 8; ++r) {
      int row = m0 + half * 8 + r;           // D-matrix: lanes 16-31 hold M+8
      float v = acc[t][r];
      if (j < N_) {
        float x = v + bl[j];
        lamBuf[(size_t)row * N_ + j] = 1.0f / (1.0f + __expf(-x));
      } else {
        buBuf[(size_t)row * N_ + (j - N_)] = v;
      }
    }
  }
}

// --------------------------- GEMM 3: output --------------------------------
// y[m, d] = sum_n H[m,n] * Wc[d,n] + u[m,d] * Dp[d].  M=16384, J=1024, K=256.
__global__ void __launch_bounds__(256)
gemm_out(const uint16_t* __restrict__ Hbf,   // (M_, N_)  bf16
         const uint16_t* __restrict__ Wcb,   // (D_, N_)  bf16
         const float*    __restrict__ u,     // (M_, D_)
         const float*    __restrict__ Dp,    // (D_)
         float* __restrict__ y)              // (M_, D_)
{
  GEMM_MAINLOOP(Hbf, Wcb, N_)

#pragma unroll
  for (int t = 0; t < 4; ++t) {
    int d = j0 + t * 16 + mr;
#pragma unroll
    for (int r = 0; r < 8; ++r) {
      int row = m0 + half * 8 + r;
      y[(size_t)row * D_ + d] = acc[t][r] + u[(size_t)row * D_ + d] * Dp[d];
    }
  }
}

// ------------------------------- scan --------------------------------------
// Pass A: per-chunk local scan, in place:
//   lamBuf[t] <- cumprod of lam within chunk (inclusive)
//   buBuf[t]  <- local h (chunk entry state = 0)
__global__ void __launch_bounds__(256)
scan_local(float* __restrict__ lamBuf, float* __restrict__ buBuf) {
  const int b = blockIdx.x / NCH;
  const int c = blockIdx.x % NCH;
  const int n = threadIdx.x;             // 256 channels, coalesced
  float h = 0.0f, p = 1.0f;
  size_t idx = ((size_t)(b * T_ + c * CHUNK)) * N_ + n;
  for (int s = 0; s < CHUNK; ++s, idx += N_) {
    float lam = lamBuf[idx];
    float bu  = buBuf[idx];
    p *= lam;
    h = lam * h + bu;
    lamBuf[idx] = p;
    buBuf[idx]  = h;
  }
}

// Pass B: sequential carry across the 32 chunks (1024 independent chains).
__global__ void __launch_bounds__(256)
scan_carry(const float* __restrict__ cumP, const float* __restrict__ Hloc,
           float* __restrict__ carry) {               // (B_, NCH, N_)
  int gid = blockIdx.x * blockDim.x + threadIdx.x;
  if (gid >= B_ * N_) return;
  int b = gid / N_, n = gid % N_;
  float h = 0.0f;
  for (int c = 0; c < NCH; ++c) {
    carry[((size_t)b * NCH + c) * N_ + n] = h;        // chunk entry state
    size_t last = ((size_t)(b * T_ + c * CHUNK + CHUNK - 1)) * N_ + n;
    h = cumP[last] * h + Hloc[last];
  }
}

// Pass C: H[t] = Hloc[t] + cumP[t] * carry(chunk(t)); emit bf16 for GEMM3.
__global__ void __launch_bounds__(256)
finalize_H(const float* __restrict__ cumP, const float* __restrict__ Hloc,
           const float* __restrict__ carry, uint16_t* __restrict__ Hbf) {
  size_t i = (size_t)blockIdx.x * blockDim.x + threadIdx.x;
  if (i >= (size_t)M_ * N_) return;
  int    n   = (int)(i % N_);
  size_t row = i / N_;                   // b*T_ + t
  int    b   = (int)(row / T_);
  int    t   = (int)(row % T_);
  int    c   = t / CHUNK;
  float h = Hloc[i] + cumP[i] * carry[((size_t)b * NCH + c) * N_ + n];
  Hbf[i] = f2bf(h);
}

// ------------------------------ launcher -----------------------------------

extern "C" void kernel_launch(void* const* d_in, const int* in_sizes, int n_in,
                              void* d_out, int out_size, void* d_ws, size_t ws_size,
                              hipStream_t stream) {
  const float* u  = (const float*)d_in[0];  // (B,T,D)
  const float* Wl = (const float*)d_in[1];  // (N,D)
  const float* bl = (const float*)d_in[2];  // (N,)
  const float* Wb = (const float*)d_in[3];  // (N,D)
  const float* Wc = (const float*)d_in[4];  // (D,N)
  const float* Dp = (const float*)d_in[5];  // (D,)
  float* y = (float*)d_out;

  char* ws = (char*)d_ws;
  size_t off = 0;
  auto alloc = [&](size_t bytes) -> void* {
    void* p = ws + off;
    off = (off + bytes + 255) & ~(size_t)255;
    return p;
  };
  uint16_t* u_bf  = (uint16_t*)alloc((size_t)M_ * D_ * 2);     // 33.5 MB
  uint16_t* Wg    = (uint16_t*)alloc((size_t)JG_ * D_ * 2);    //  1.0 MB
  uint16_t* Wcb   = (uint16_t*)alloc((size_t)D_ * N_ * 2);     //  0.5 MB
  float*    lamB  = (float*)   alloc((size_t)M_ * N_ * 4);     // 16.8 MB
  float*    buB   = (float*)   alloc((size_t)M_ * N_ * 4);     // 16.8 MB
  float*    carry = (float*)   alloc((size_t)B_ * NCH * N_ * 4);
  uint16_t* Hbf   = (uint16_t*)alloc((size_t)M_ * N_ * 2);     //  8.4 MB

  // prep: bf16 operands
  cvt_f32_bf16<<<(M_ * D_ + 255) / 256, 256, 0, stream>>>(u, u_bf, M_ * D_);
  prep_wg<<<(JG_ * D_ + 255) / 256, 256, 0, stream>>>(Wl, Wb, Wg);
  cvt_f32_bf16<<<(D_ * N_ + 255) / 256, 256, 0, stream>>>(Wc, Wcb, D_ * N_);

  // GEMM1: gates (lam, Bu) fused
  gemm_gates<<<dim3(M_ / 32, JG_ / 256), 256, 0, stream>>>(u_bf, Wg, bl, lamB, buB);

  // chunked linear-recurrence scan
  scan_local<<<B_ * NCH, 256, 0, stream>>>(lamB, buB);
  scan_carry<<<(B_ * N_ + 255) / 256, 256, 0, stream>>>(lamB, buB, carry);
  finalize_H<<<(int)(((size_t)M_ * N_ + 255) / 256), 256, 0, stream>>>(lamB, buB, carry, Hbf);

  // GEMM3: output projection + D*u residual
  gemm_out<<<dim3(M_ / 32, D_ / 256), 256, 0, stream>>>(Hbf, Wcb, u, Dp, y);
}
